// FlashAttention_50130858279276
// MI455X (gfx1250) — matrix operand order repaired and checked by reading, not compiled, not run on hardware
//
#include <hip/hip_runtime.h>

typedef _Float16 h8  __attribute__((ext_vector_type(8)));
typedef _Float16 h16 __attribute__((ext_vector_type(16)));
typedef float    f8  __attribute__((ext_vector_type(8)));

static __device__ __forceinline__ f8 wmma_f16(h16 a, h16 b, f8 c) {
  return __builtin_amdgcn_wmma_f32_16x16x32_f16(false, a, false, b, (short)0, c, false, false);
}

static __device__ __forceinline__ h16 pack16(h8 lo, h8 hi) {
  h16 r;
#pragma unroll
  for (int i = 0; i < 8; ++i) { r[i] = lo[i]; r[i + 8] = hi[i]; }
  return r;
}

// A fragment: 16x32 f16 tile (rows row0..row0+15, K = kk..kk+31), row-major src, ld elems.
// ISA layout: lane half h: frag[0..7] = K(kk+8h..), frag[8..15] = K(kk+16+8h..)
static __device__ __forceinline__ h16 load_a16(const _Float16* __restrict__ base, int ld,
                                               int row0, int kk, int lane) {
  const int m = lane & 15, h = lane >> 4;
  const _Float16* p = base + (size_t)(row0 + m) * ld + kk + h * 8;
  return pack16(*(const h8*)p, *(const h8*)(p + 16));
}

// B fragment: 32x16 f16 tile (cols col0..col0+15, K = kk..kk+31) where source stores B^T
// row-major (row = output col, K contiguous). ISA layout: lane half h: frag[0..7] = K(kk+8h..), frag[8..15] = K(kk+16+8h..)
static __device__ __forceinline__ h16 load_b16(const _Float16* __restrict__ base, int ld,
                                               int col0, int kk, int lane) {
  const int n = lane & 15, h = lane >> 4;
  const _Float16* p = base + (size_t)(col0 + n) * ld + kk + h * 8;
  return pack16(*(const h8*)p, *(const h8*)(p + 16));
}

__global__ __launch_bounds__(256) void cvt_f32_to_f16_kernel(const float* __restrict__ in,
                                                             _Float16* __restrict__ out, int n) {
  int i = blockIdx.x * 256 + threadIdx.x;
  if (i < n) out[i] = (_Float16)in[i];
}

// ---------------- QKV projection: [4096,1024] x [3072,1024]^T -> Q,K head-major, V transposed
__global__ __launch_bounds__(128) void qkv_gemm_kernel(const _Float16* __restrict__ X,
                                                       const _Float16* __restrict__ W,
                                                       _Float16* __restrict__ Qd,
                                                       _Float16* __restrict__ Kd,
                                                       _Float16* __restrict__ VT) {
  const int lane = threadIdx.x & 31;
  const int wv   = threadIdx.x >> 5;
  const int row0 = blockIdx.y * 64 + (wv >> 1) * 32;
  const int col0 = blockIdx.x * 128 + (wv & 1) * 64;

  f8 acc[2][4] = {};
  for (int kk = 0; kk < 1024; kk += 32) {
    h16 af[2], bf[4];
    af[0] = load_a16(X, 1024, row0,      kk, lane);
    af[1] = load_a16(X, 1024, row0 + 16, kk, lane);
#pragma unroll
    for (int j = 0; j < 4; ++j) bf[j] = load_b16(W, 1024, col0 + j * 16, kk, lane);
#pragma unroll
    for (int i = 0; i < 2; ++i)
#pragma unroll
      for (int j = 0; j < 4; ++j) acc[i][j] = wmma_f16(af[i], bf[j], acc[i][j]);
  }

  const int h = lane >> 4, n = lane & 15;
#pragma unroll
  for (int i = 0; i < 2; ++i)
#pragma unroll
    for (int j = 0; j < 4; ++j)
#pragma unroll
      for (int v = 0; v < 8; ++v) {
        const int r = row0 + i * 16 + 8 * h + v;   // token row in [0,4096)
        const int c = col0 + j * 16 + n;           // output col in [0,3072)
        const int b = r >> 11, s = r & 2047;
        const int which = c >> 10, cc = c & 1023, head = cc >> 6, d = cc & 63;
        const size_t bh = (size_t)(b * 16 + head);
        const _Float16 val = (_Float16)acc[i][j][v];
        if (which == 0)      Qd[(bh * 2048 + s) * 64 + d] = val;
        else if (which == 1) Kd[(bh * 2048 + s) * 64 + d] = val;
        else                 VT[(bh * 64 + d) * 2048 + s] = val;  // V transposed [hd, S]
      }
}

// ---------------- Flash attention: one wave handles 16 query rows of one (b, head)
__global__ __launch_bounds__(128) void flash_attn_kernel(const _Float16* __restrict__ Qd,
                                                         const _Float16* __restrict__ Kd,
                                                         const _Float16* __restrict__ VT,
                                                         _Float16* __restrict__ Od) {
  __shared__ __align__(16) _Float16 Pbuf[4][16 * 32];  // 1KB per wave staging for C->A relayout
  const int lane = threadIdx.x & 31;
  const int wv   = threadIdx.x >> 5;
  const int gw   = blockIdx.x * 4 + wv;      // 0..4095
  const int qt   = gw & 127;
  const int head = (gw >> 7) & 15;
  const int b    = gw >> 11;
  const size_t bh = (size_t)(b * 16 + head);

  const _Float16* Qb = Qd + bh * 2048 * 64;
  const _Float16* Kb = Kd + bh * 2048 * 64;
  const _Float16* Vb = VT + bh * 64 * 2048;
  const int q0 = qt * 16;
  const int h = lane >> 4, n = lane & 15;
  _Float16* P = Pbuf[wv];

  const h16 aq0 = load_a16(Qb, 64, q0, 0, lane);
  const h16 aq1 = load_a16(Qb, 64, q0, 32, lane);

  f8 o[4] = {};
  float mrow[8], lrow[8];
#pragma unroll
  for (int v = 0; v < 8; ++v) { mrow[v] = -INFINITY; lrow[v] = 0.0f; }

  for (int kc = 0; kc < 2048; kc += 32) {
    // scores: two 16x16 tiles, hd=64 -> two K-steps each
    f8 s0 = {}, s1 = {};
    s0 = wmma_f16(aq0, load_b16(Kb, 64, kc,      0,  lane), s0);
    s0 = wmma_f16(aq1, load_b16(Kb, 64, kc,      32, lane), s0);
    s1 = wmma_f16(aq0, load_b16(Kb, 64, kc + 16, 0,  lane), s1);
    s1 = wmma_f16(aq1, load_b16(Kb, 64, kc + 16, 32, lane), s1);

#pragma unroll
    for (int v = 0; v < 8; ++v) {
      const float x0 = s0[v] * 0.125f;   // 1/sqrt(64)
      const float x1 = s1[v] * 0.125f;
      float cm = fmaxf(x0, x1);
      cm = fmaxf(cm, __shfl_xor(cm, 1, 32));
      cm = fmaxf(cm, __shfl_xor(cm, 2, 32));
      cm = fmaxf(cm, __shfl_xor(cm, 4, 32));
      cm = fmaxf(cm, __shfl_xor(cm, 8, 32));
      const float mnew  = fmaxf(mrow[v], cm);
      const float alpha = __expf(mrow[v] - mnew);
      const float p0 = __expf(x0 - mnew);
      const float p1 = __expf(x1 - mnew);
      float rs = p0 + p1;
      rs += __shfl_xor(rs, 1, 32);
      rs += __shfl_xor(rs, 2, 32);
      rs += __shfl_xor(rs, 4, 32);
      rs += __shfl_xor(rs, 8, 32);
      lrow[v] = lrow[v] * alpha + rs;
      mrow[v] = mnew;
#pragma unroll
      for (int t = 0; t < 4; ++t) o[t][v] *= alpha;
      const int prow = v + 8 * h;               // C-layout row for this lane/VGPR
      P[prow * 32 + n]      = (_Float16)p0;     // chunk cols 0..15
      P[prow * 32 + 16 + n] = (_Float16)p1;     // chunk cols 16..31
    }
    __builtin_amdgcn_wave_barrier();            // keep DS stores before DS loads (in-order per wave)

    // reload P as A fragment (16 queries x 32 keys)
    const _Float16* pr = P + (lane & 15) * 32 + h * 8;
    const h16 pf = pack16(*(const h8*)pr, *(const h8*)(pr + 16));
    __builtin_amdgcn_wave_barrier();

    // O += P * V : V^T rows are key-contiguous -> direct B fragments
#pragma unroll
    for (int t = 0; t < 4; ++t)
      o[t] = wmma_f16(pf, load_b16(Vb, 2048, t * 16, kc, lane), o[t]);
  }

  // write attn output [B,S,H] f16 (A operand of output GEMM)
#pragma unroll
  for (int t = 0; t < 4; ++t)
#pragma unroll
    for (int v = 0; v < 8; ++v) {
      const int s = q0 + v + 8 * h;
      const int c = head * 64 + t * 16 + n;
      Od[((size_t)b * 2048 + s) * 1024 + c] = (_Float16)(o[t][v] / lrow[v]);
    }
}

// ---------------- output projection: [4096,1024] x [1024,1024]^T -> f32 out
__global__ __launch_bounds__(128) void out_gemm_kernel(const _Float16* __restrict__ A,
                                                       const _Float16* __restrict__ W,
                                                       float* __restrict__ out) {
  const int lane = threadIdx.x & 31;
  const int wv   = threadIdx.x >> 5;
  const int row0 = blockIdx.y * 64 + (wv >> 1) * 32;
  const int col0 = blockIdx.x * 128 + (wv & 1) * 64;

  f8 acc[2][4] = {};
  for (int kk = 0; kk < 1024; kk += 32) {
    h16 af[2], bf[4];
    af[0] = load_a16(A, 1024, row0,      kk, lane);
    af[1] = load_a16(A, 1024, row0 + 16, kk, lane);
#pragma unroll
    for (int j = 0; j < 4; ++j) bf[j] = load_b16(W, 1024, col0 + j * 16, kk, lane);
#pragma unroll
    for (int i = 0; i < 2; ++i)
#pragma unroll
      for (int j = 0; j < 4; ++j) acc[i][j] = wmma_f16(af[i], bf[j], acc[i][j]);
  }

  const int h = lane >> 4, n = lane & 15;
#pragma unroll
  for (int i = 0; i < 2; ++i)
#pragma unroll
    for (int j = 0; j < 4; ++j)
#pragma unroll
      for (int v = 0; v < 8; ++v) {
        const int r = row0 + i * 16 + 8 * h + v;
        const int c = col0 + j * 16 + n;
        out[(size_t)r * 1024 + c] = acc[i][j][v];
      }
}

extern "C" void kernel_launch(void* const* d_in, const int* in_sizes, int n_in,
                              void* d_out, int out_size, void* d_ws, size_t ws_size,
                              hipStream_t stream) {
  const float* X    = (const float*)d_in[0];  // [2,2048,1024]
  const float* Wqkv = (const float*)d_in[1];  // [3072,1024]
  const float* Wo   = (const float*)d_in[2];  // [1024,1024]
  float* out = (float*)d_out;                 // [2,2048,1024] f32

  const int NX = 4096 * 1024;       // 4,194,304
  const int NW1 = 3072 * 1024;      // 3,145,728
  const int NW2 = 1024 * 1024;      // 1,048,576
  const int NQ = 2 * 16 * 2048 * 64;// 4,194,304 per Q/K/V

  _Float16* ws = (_Float16*)d_ws;
  _Float16* Xh   = ws;                 // 4,194,304
  _Float16* W1h  = Xh  + NX;           // 3,145,728
  _Float16* W2h  = W1h + NW1;          // 1,048,576
  _Float16* Qd   = W2h + NW2;          // 4,194,304   [b,h,S,hd]
  _Float16* Kd   = Qd  + NQ;           // 4,194,304   [b,h,S,hd]
  _Float16* VT   = Kd  + NQ;           // 4,194,304   [b,h,hd,S]
  _Float16* Attn = VT  + NQ;           // 4,194,304   [B,S,H]

  cvt_f32_to_f16_kernel<<<(NX + 255) / 256, 256, 0, stream>>>(X, Xh, NX);
  cvt_f32_to_f16_kernel<<<(NW1 + 255) / 256, 256, 0, stream>>>(Wqkv, W1h, NW1);
  cvt_f32_to_f16_kernel<<<(NW2 + 255) / 256, 256, 0, stream>>>(Wo, W2h, NW2);

  qkv_gemm_kernel<<<dim3(3072 / 128, 4096 / 64), 128, 0, stream>>>(Xh, W1h, Qd, Kd, VT);
  flash_attn_kernel<<<4096 / 4, 128, 0, stream>>>(Qd, Kd, VT, Attn);
  out_gemm_kernel<<<dim3(1024 / 128, 4096 / 64), 128, 0, stream>>>(Attn, W2h, out);
}
